// SelfAttention_46394236731496
// MI455X (gfx1250) — matrix operand order repaired and checked by reading, not compiled, not run on hardware
//
#include <hip/hip_runtime.h>
#include <hip/hip_bf16.h>
#include <math.h>

// ---------------------------------------------------------------------------
// Self-attention (B=8,T=1024,E=128,H=8) for gfx1250.
// v_wmma_f32_16x16x32_bf16 for all GEMMs; global_load_async_to_lds_b128 for
// Q-tile staging; bf16 P staged in LDS for the P*V GEMM.
// ---------------------------------------------------------------------------

typedef __bf16 bf16;
typedef __attribute__((ext_vector_type(16))) __bf16 v16bf;
typedef __attribute__((ext_vector_type(8)))  float  v8f;

#define WMMA_BF16(A, B, C) \
    __builtin_amdgcn_wmma_f32_16x16x32_bf16(false, (A), false, (B), (short)0, (C), false, false)

// ---- fragment builders (contiguous k order, same for A and B) ---------------------
// A: 16x32 (MxK), row-major source (works for global or LDS pointers).
// lanes 0-15: row=lane, elems 0..15 = K[k0..k0+15]
// lanes 16-31: row=lane-16, elems 0..15 = K[k0+16..k0+31].
__device__ __forceinline__ v16bf load_a_frag(const bf16* __restrict__ src, int ldm,
                                             int row0, int k0, int lane) {
    int row = row0 + (lane & 15);
    int kb  = k0 + ((lane & 16) ? 16 : 0);
    const bf16* p = src + row * ldm + kb;
    v16bf a;
#pragma unroll
    for (int j = 0; j < 16; ++j) a[j] = p[j];
    return a;
}

// B: 32x16 (KxN) from an N-major source srcT[n][k] (contiguous K per column).
// lanes 0-15: col=lane, K = k0..k0+15 ; lanes 16-31: col=lane-16, K = k0+16..k0+31
__device__ __forceinline__ v16bf load_b_frag(const bf16* __restrict__ srcT, int ldk,
                                             int n0, int k0, int lane) {
    int n  = n0 + (lane & 15);
    int kb = k0 + ((lane & 16) ? 16 : 0);
    const bf16* p = srcT + n * ldk + kb;
    v16bf b;
#pragma unroll
    for (int j = 0; j < 16; ++j) b[j] = p[j];
    return b;
}

// ---- CDNA5 async global->LDS copy (ASYNCcnt path) -------------------------
// Copies 16 bytes from global to LDS; tracked by ASYNCcnt.
__device__ __forceinline__ void async_copy_b128(void* lds_dst, const void* gsrc) {
    unsigned int loff = (unsigned int)(size_t)lds_dst;   // low 32 bits of shared
                                                         // aperture = LDS offset
    asm volatile("global_load_async_to_lds_b128 %0, %1, off"
                 :: "v"(loff), "v"(gsrc)
                 : "memory");
}
__device__ __forceinline__ void wait_asynccnt0() {
    asm volatile("s_wait_asynccnt 0x0" ::: "memory");
}

// ---- prep kernels ----------------------------------------------------------
__global__ void cvt_x_kernel(const float* __restrict__ x, bf16* __restrict__ xbf, int n) {
    int i = blockIdx.x * blockDim.x + threadIdx.x;
    if (i < n) xbf[i] = (bf16)x[i];
}

// W: (E=128, N=1024) row-major f32  ->  WT: (N=1024, E=128) bf16 (N-major)
__global__ void transpose_w_kernel(const float* __restrict__ Wq, const float* __restrict__ Wk,
                                   const float* __restrict__ Wv,
                                   bf16* __restrict__ WTq, bf16* __restrict__ WTk,
                                   bf16* __restrict__ WTv) {
    int which = blockIdx.z;
    const float* W = (which == 0) ? Wq : (which == 1) ? Wk : Wv;
    bf16* WT       = (which == 0) ? WTq : (which == 1) ? WTk : WTv;
    int i = blockIdx.x * blockDim.x + threadIdx.x;   // 0..131071, n = i>>7, e = i&127
    int n = i >> 7, e = i & 127;
    WT[i] = (bf16)W[e * 1024 + n];
}

__global__ void zero_kernel(float* __restrict__ p, int n) {
    int i = blockIdx.x * blockDim.x + threadIdx.x;
    if (i < n) p[i] = 0.0f;
}

// ---- QKV projection: Y = x @ W + b, split heads, bf16 out -----------------
// grid: (1024,1,3) blocks of 256 (8 waves); 1 wave = one 16x64 output strip
// (4 accumulators, each A-fragment reused 4x).
__global__ void proj_kernel(const bf16* __restrict__ xbf,
                            const bf16* __restrict__ WTq, const bf16* __restrict__ WTk,
                            const bf16* __restrict__ WTv,
                            const float* __restrict__ bq, const float* __restrict__ bk,
                            const float* __restrict__ bv,
                            bf16* __restrict__ Qb, bf16* __restrict__ Kb,
                            bf16* __restrict__ Vt) {
    int which = blockIdx.z;
    const bf16* WT   = (which == 0) ? WTq : (which == 1) ? WTk : WTv;
    const float* bia = (which == 0) ? bq  : (which == 1) ? bk  : bv;

    int lane = threadIdx.x & 31, wave = threadIdx.x >> 5;
    int job = blockIdx.x * 8 + wave;     // 0..8191 = 512 mtiles * 16 ngroups
    int mt = job >> 4, ng = job & 15;
    int m0 = mt * 16, n0 = ng * 64;

    v8f acc[4] = {};
#pragma unroll
    for (int kc = 0; kc < 4; ++kc) {
        v16bf a = load_a_frag(xbf, 128, m0, kc * 32, lane);
#pragma unroll
        for (int g = 0; g < 4; ++g) {
            v16bf b = load_b_frag(WT, 128, n0 + g * 16, kc * 32, lane);
            acc[g] = WMMA_BF16(a, b, acc[g]);
        }
    }

    const float norm = 0.08838834764831845f;   // 1/sqrt(128)
#pragma unroll
    for (int g = 0; g < 4; ++g) {
        int col = n0 + g * 16 + (lane & 15);
        int h = col >> 7, e = col & 127;
        float bval = bia[col];
#pragma unroll
        for (int r = 0; r < 8; ++r) {
            int gm = m0 + r + ((lane & 16) ? 8 : 0);   // 0..8191
            int bb = gm >> 10, t = gm & 1023;
            int bh = bb * 8 + h;
            float val = acc[g][r] + bval;
            if (which == 0)      Qb[(bh * 1024 + t) * 128 + e] = (bf16)(val * norm);
            else if (which == 1) Kb[(bh * 1024 + t) * 128 + e] = (bf16)(val * norm);
            else                 Vt[(bh * 128 + e) * 1024 + t] = (bf16)val;   // transposed
        }
    }
}

// ---- attention core --------------------------------------------------------
// grid: (64 heads, 8 q-groups), 256 threads (8 waves). Each block: 8 q-tiles.
// Per q-tile: async-stage Q tile to LDS, S = Q*K^T (WMMA -> LDS f32),
// masked softmax (normalized P written as bf16 to LDS), O = P*V (WMMA),
// row-sums accumulated into headAcc[bh*128+e] via global f32 atomics.
#define LDP   1032   // sS stride (f32): +8 floats -> 8-bank shift, 32B aligned rows
#define LDPB  1040   // sP stride (bf16): +16 -> 8-bank shift, 16B aligned rows
#define LDQ   136    // sQ stride (bf16): +8 -> 4-bank shift, 16B aligned rows

__global__ void attn_kernel(const bf16* __restrict__ Qb, const bf16* __restrict__ Kb,
                            const bf16* __restrict__ Vt, const float* __restrict__ mask,
                            float* __restrict__ headAcc) {
    extern __shared__ char smem[];
    float* sS   = (float*)smem;                              // 16*LDP f32   (66048 B)
    bf16*  sP   = (bf16*)(smem + 16 * LDP * 4);              // 16*LDPB bf16 (33280 B)
    bf16*  sQ   = (bf16*)(smem + 16 * LDP * 4 + 16 * LDPB * 2);          // 16*LDQ bf16
    float* sMax = (float*)(smem + 16 * LDP * 4 + 16 * LDPB * 2 + 16 * LDQ * 2); // 16
    float* sSum = sMax + 16;                                 // 16
    float* sRed = sSum + 16;                                 // 256

    int bh = blockIdx.x;
    int qg = blockIdx.y;
    int tid = threadIdx.x;
    int lane = tid & 31, wave = tid >> 5;

    const bf16* Qh = Qb + (size_t)bh * 1024 * 128;
    const bf16* Kh = Kb + (size_t)bh * 1024 * 128;
    const bf16* Vh = Vt + (size_t)bh * 128 * 1024;

    int et = wave;            // this wave's 16-wide e-tile for the PV GEMM
    float osum = 0.0f;        // per-lane running sum over q rows for column e

    for (int qi = 0; qi < 8; ++qi) {
        int qt = qg * 8 + qi;
        int q0 = qt * 16;

        // ---- Stage Q tile (16x128 bf16 = 4KB) into LDS asynchronously ----
        {
            int row = tid >> 4, ch = tid & 15;               // 256 x 16B chunks
            const bf16* g = Qh + (size_t)(q0 + row) * 128 + ch * 8;
            bf16* d = sQ + row * LDQ + ch * 8;
            async_copy_b128(d, g);
        }
        wait_asynccnt0();
        __syncthreads();

        // ---- Phase 1: S(16x1024) = Q_tile * K^T ----
        v16bf afr[4];
#pragma unroll
        for (int kc = 0; kc < 4; ++kc) afr[kc] = load_a_frag(sQ, LDQ, 0, kc * 32, lane);

        for (int i = 0; i < 8; ++i) {
            int nt = wave * 8 + i;                      // key tile 0..63
            if (i + 1 < 8)                               // prefetch next K tile row
                __builtin_prefetch(Kh + (size_t)(nt + 1) * 16 * 128 + (lane & 15) * 128, 0, 3);
            v8f acc = {};
#pragma unroll
            for (int kc = 0; kc < 4; ++kc) {
                v16bf b = load_b_frag(Kh, 128, nt * 16, kc * 32, lane);
                acc = WMMA_BF16(afr[kc], b, acc);
            }
            int coln = nt * 16 + (lane & 15);
#pragma unroll
            for (int r = 0; r < 8; ++r) {
                int row = r + ((lane & 16) ? 8 : 0);
                sS[row * LDP + coln] = acc[r];
            }
        }
        __syncthreads();

        // ---- Phase 2: masked softmax over 1024 keys; write bf16 P ----
        {
            int r   = tid >> 4;       // 0..15
            int sub = tid & 15;
            float mrow = mask[bh * 1024 + q0 + r];

            float mx = -3.0e38f;
            for (int c = sub; c < 1024; c += 16) {
                float s  = sS[r * LDP + c];
                float se = (mrow == 0.0f || s == 0.0f) ? -1.0e9f : s;
                mx = fmaxf(mx, se);
            }
            sRed[tid] = mx;
            __syncthreads();
            if (sub == 0) {
                float m = sRed[r * 16];
#pragma unroll
                for (int j = 1; j < 16; ++j) m = fmaxf(m, sRed[r * 16 + j]);
                sMax[r] = m;
            }
            __syncthreads();
            float m = sMax[r];
            float ls = 0.0f;
            for (int c = sub; c < 1024; c += 16) {
                float s  = sS[r * LDP + c];
                float se = (mrow == 0.0f || s == 0.0f) ? -1.0e9f : s;
                float p  = __expf(se - m);
                sS[r * LDP + c] = p;
                ls += p;
            }
            sRed[tid] = ls;
            __syncthreads();
            if (sub == 0) {
                float t = 0.0f;
#pragma unroll
                for (int j = 0; j < 16; ++j) t += sRed[r * 16 + j];
                sSum[r] = t;
            }
            __syncthreads();
            float inv = 1.0f / sSum[r];
            for (int c = sub; c < 1024; c += 16)
                sP[r * LDPB + c] = (bf16)(sS[r * LDP + c] * inv);
        }
        __syncthreads();

        // ---- Phase 3: O(16x16 e-tile) = P(16x1024) * V(1024x16) ----
        {
            v8f acc = {};
            for (int kc = 0; kc < 32; ++kc) {
                v16bf a = load_a_frag(sP, LDPB, 0, kc * 32, lane);   // bf16 direct
                v16bf b = load_b_frag(Vh, 1024, et * 16, kc * 32, lane);
                acc = WMMA_BF16(a, b, acc);
            }
            float part = 0.0f;
#pragma unroll
            for (int r = 0; r < 8; ++r) part += acc[r];   // sum this lane's 8 q-rows
            osum += part;
        }
        __syncthreads();
    }

    int e = et * 16 + (lane & 15);
    atomicAdd(&headAcc[bh * 128 + e], osum);   // both half-waves cover all 16 rows
}

// ---- epilogue: out = tanh(acc(8,1024) @ Wu(1024,128) + bu) ----------------
__global__ void final_kernel(const float* __restrict__ headAcc, const float* __restrict__ Wu,
                             const float* __restrict__ bu, float* __restrict__ out) {
    int b = blockIdx.x;      // 0..7
    int j = threadIdx.x;     // 0..127
    float acc = bu[j];
    for (int n = 0; n < 1024; ++n)
        acc += headAcc[b * 1024 + n] * Wu[n * 128 + j];
    out[b * 128 + j] = tanhf(acc);
}

// ---------------------------------------------------------------------------
extern "C" void kernel_launch(void* const* d_in, const int* in_sizes, int n_in,
                              void* d_out, int out_size, void* d_ws, size_t ws_size,
                              hipStream_t stream) {
    const float* x    = (const float*)d_in[0];
    const float* mask = (const float*)d_in[1];
    const float* Wq   = (const float*)d_in[2];
    const float* bq   = (const float*)d_in[3];
    const float* Wk   = (const float*)d_in[4];
    const float* bk   = (const float*)d_in[5];
    const float* Wv   = (const float*)d_in[6];
    const float* bv   = (const float*)d_in[7];
    const float* Wu   = (const float*)d_in[8];
    const float* bu   = (const float*)d_in[9];
    float* out = (float*)d_out;

    char* ws = (char*)d_ws;
    size_t off = 0;
    auto alloc = [&](size_t bytes) -> void* {
        void* p = ws + off;
        off = (off + bytes + 255) & ~(size_t)255;
        return p;
    };
    bf16*  xbf     = (bf16*)alloc((size_t)8192 * 128 * 2);        // x in bf16
    bf16*  WTq     = (bf16*)alloc((size_t)1024 * 128 * 2);        // W^T bf16
    bf16*  WTk     = (bf16*)alloc((size_t)1024 * 128 * 2);
    bf16*  WTv     = (bf16*)alloc((size_t)1024 * 128 * 2);
    bf16*  Qb      = (bf16*)alloc((size_t)64 * 1024 * 128 * 2);   // (bh,t,e)
    bf16*  Kb      = (bf16*)alloc((size_t)64 * 1024 * 128 * 2);   // (bh,t,e)
    bf16*  Vt      = (bf16*)alloc((size_t)64 * 128 * 1024 * 2);   // (bh,e,t)
    float* headAcc = (float*)alloc((size_t)64 * 128 * 4);         // per-head col sums

    // prep
    cvt_x_kernel<<<4096, 256, 0, stream>>>(x, xbf, 8192 * 128);
    transpose_w_kernel<<<dim3(512, 1, 3), 256, 0, stream>>>(Wq, Wk, Wv, WTq, WTk, WTv);
    zero_kernel<<<32, 256, 0, stream>>>(headAcc, 64 * 128);

    // QKV projections (WMMA, 16x64 strip per wave)
    proj_kernel<<<dim3(1024, 1, 3), 256, 0, stream>>>(xbf, WTq, WTk, WTv,
                                                      bq, bk, bv, Qb, Kb, Vt);

    // attention (WMMA + async LDS staging) : 64 heads x 8 q-groups
    size_t smem = (size_t)16 * LDP * 4        // sS  f32
                + (size_t)16 * LDPB * 2       // sP  bf16
                + (size_t)16 * LDQ * 2        // sQ  bf16
                + (size_t)(16 + 16 + 256) * 4;
    attn_kernel<<<dim3(64, 8), 256, smem, stream>>>(Qb, Kb, Vt, mask, headAcc);

    // epilogue
    final_kernel<<<8, 128, 0, stream>>>(headAcc, Wu, bu, out);
}
